// TOGL_67276367725142
// MI455X (gfx1250) — matrix operand order repaired and checked by reading, not compiled, not run on hardware
//
#include <hip/hip_runtime.h>
#include <hip/hip_bf16.h>

// ---------------------------------------------------------------------------
// TOGL pipeline for MI455X (gfx1250, wave32, WMMA).
//   filt   = relu(X @ W1^T + b1) @ W2^T + b2        (WMMA bf16, bias in C)
//   pos    = stable rank of filt[:,f]               (LDS-tiled rank count)
//   pd     = 0-dim persistence (sequential UF, elder rule) in LDS, 4 blocks
//   pooled = pd . colsum(Wt) + sum(bt)              (DeepSets sum collapsed)
//   out    = X + pooled * Wr + br                   (streaming float4 pass)
// ---------------------------------------------------------------------------

#define N_NODES 8192
#define N_FEAT  1024
#define MAX_DEG 8
#define N_FILT  4

typedef __attribute__((ext_vector_type(16))) __bf16 v16bf;
typedef __attribute__((ext_vector_type(8)))  float  v8f;
typedef __attribute__((ext_vector_type(4)))  float  v4f;

// ---------------------------------------------------------------------------
// Kernel 1: filt = relu(X @ W1^T + b1) @ W2^T + b2
// 128 threads = 4 waves, each wave owns a 16-row tile (64 rows / block).
// GEMM1: two 16x16 f32 accumulators (N=0..15, 16..31), K-loop of 32 over 1024.
// GEMM2: relu(h) staged through LDS -> A-frag, W2 as sparse B (cols 0..3).
// ---------------------------------------------------------------------------
__global__ __launch_bounds__(128) void togl_filt_kernel(
    const float* __restrict__ X, const float* __restrict__ W1,
    const float* __restrict__ b1, const float* __restrict__ W2,
    const float* __restrict__ b2, float* __restrict__ filt)
{
    __shared__ float lds_h[4][16][33];   // [wave][M][N], padded stride

    const int lane = threadIdx.x & 31;
    const int wave = threadIdx.x >> 5;
    const int l    = lane & 15;
    const int half = lane >> 4;          // K-half selector per WMMA layout
    const int m0   = blockIdx.x * 64 + wave * 16;

    // C init = bias broadcast (acc0 covers N=l, acc1 covers N=16+l)
    v8f acc0, acc1;
    {
        const float bn0 = b1[l];
        const float bn1 = b1[16 + l];
#pragma unroll
        for (int r = 0; r < 8; ++r) { acc0[r] = bn0; acc1[r] = bn1; }
    }

    const float* __restrict__ rowX = X + (size_t)(m0 + l) * N_FEAT;
    const int baseKA = half * 16;        // A-frag lane K base: 0 or 16

    for (int k0 = 0; k0 < N_FEAT; k0 += 32) {
        // ---- A fragment: 16x32 bf16, lane l holds row M=l ----
        const float* pA = rowX + k0 + baseKA;
        __builtin_prefetch(pA + 32, 0, 0);           // global_prefetch next tile
        v4f a0 = *(const v4f*)(pA + 0);
        v4f a1 = *(const v4f*)(pA + 4);
        v4f a2 = *(const v4f*)(pA + 8);
        v4f a3 = *(const v4f*)(pA + 12);
        v16bf A;
#pragma unroll
        for (int q = 0; q < 4; ++q) {
            A[q]      = (__bf16)a0[q];
            A[4 + q]  = (__bf16)a1[q];
            A[8 + q]  = (__bf16)a2[q];
            A[12 + q] = (__bf16)a3[q];
        }

        // ---- B fragments from W1: B[k][n] = W1[n][k0+k]; lane holds col n ----
        const float* pB0 = W1 + (size_t)l * N_FEAT + k0 + half * 16;
        const float* pB1 = pB0 + 16 * N_FEAT;
        v16bf B0, B1;
#pragma unroll
        for (int e = 0; e < 16; e += 4) {
            v4f t0 = *(const v4f*)(pB0 + e);
            v4f t1 = *(const v4f*)(pB1 + e);
#pragma unroll
            for (int q = 0; q < 4; ++q) {
                B0[e + q] = (__bf16)t0[q];
                B1[e + q] = (__bf16)t1[q];
            }
        }

        acc0 = __builtin_amdgcn_wmma_f32_16x16x32_bf16(
            false, A, false, B0, (short)0, acc0, false, false);
        acc1 = __builtin_amdgcn_wmma_f32_16x16x32_bf16(
            false, A, false, B1, (short)0, acc1, false, false);
    }

    // relu + stage h tile to LDS in [M][N] layout
#pragma unroll
    for (int r = 0; r < 8; ++r) {
        lds_h[wave][half * 8 + r][l]      = fmaxf(acc0[r], 0.0f);
        lds_h[wave][half * 8 + r][16 + l] = fmaxf(acc1[r], 0.0f);
    }
    __syncthreads();

    // ---- GEMM2: filt_tile = h(16x32) @ W2^T(32x4), padded to N=16 ----
    v16bf A2;
#pragma unroll
    for (int e = 0; e < 8; ++e) {
        A2[e]     = (__bf16)lds_h[wave][l][half * 16 + e];
        A2[8 + e] = (__bf16)lds_h[wave][l][half * 16 + 8 + e];
    }
    v16bf B2;
    if (l < N_FILT) {
        const float* pW2 = W2 + l * 32 + half * 16;
#pragma unroll
        for (int e = 0; e < 16; e += 4) {
            v4f t = *(const v4f*)(pW2 + e);
#pragma unroll
            for (int q = 0; q < 4; ++q) B2[e + q] = (__bf16)t[q];
        }
    } else {
#pragma unroll
        for (int e = 0; e < 16; ++e) B2[e] = (__bf16)0.0f;
    }
    v8f acc2;
    {
        const float bf = (l < N_FILT) ? b2[l] : 0.0f;
#pragma unroll
        for (int r = 0; r < 8; ++r) acc2[r] = bf;
    }
    acc2 = __builtin_amdgcn_wmma_f32_16x16x32_bf16(
        false, A2, false, B2, (short)0, acc2, false, false);

    if (l < N_FILT) {
#pragma unroll
        for (int r = 0; r < 8; ++r)
            filt[(size_t)(m0 + half * 8 + r) * N_FILT + l] = acc2[r];
    }
}

// ---------------------------------------------------------------------------
// Kernel 2: stable rank (== stable argsort) per filtration, LDS-tiled.
// pos[f][i] = #{u : filt[u] < filt[i]  or  (== and u < i)} ; order[f][pos]=i
// ---------------------------------------------------------------------------
__global__ __launch_bounds__(256) void togl_rank_kernel(
    const float* __restrict__ filt, int* __restrict__ pos,
    int* __restrict__ order)
{
    __shared__ float sv[256];
    const int f = blockIdx.y;
    const int i = blockIdx.x * 256 + threadIdx.x;
    const float vi = filt[(size_t)i * N_FILT + f];
    int cnt = 0;
    for (int c = 0; c < N_NODES; c += 256) {
        sv[threadIdx.x] = filt[(size_t)(c + threadIdx.x) * N_FILT + f];
        __syncthreads();
#pragma unroll 8
        for (int t = 0; t < 256; ++t) {
            const float vu = sv[t];
            const int   u  = c + t;
            cnt += (vu < vi) || (vu == vi && u < i);
        }
        __syncthreads();
    }
    pos[f * N_NODES + i]     = cnt;
    order[f * N_NODES + cnt] = i;
}

// ---------------------------------------------------------------------------
// Kernel 3: 0-dim persistence via sequential union-find, all state in LDS
// (128 KB dynamic). One block per filtration; lane 0 runs the elder-rule scan.
// pd[v][2f+0] = pos[v]; pd[v][2f+1] = death (n if undying).
// ---------------------------------------------------------------------------
extern __shared__ int uf_smem[];
__global__ void togl_uf_kernel(
    const int* __restrict__ dst_edges, const int* __restrict__ pos_g,
    const int* __restrict__ order_g, float* __restrict__ pd)
{
    const int f = blockIdx.x;
    int*   parent = uf_smem;
    int*   posL   = uf_smem + N_NODES;
    int*   ordL   = uf_smem + 2 * N_NODES;
    float* deaths = (float*)(uf_smem + 3 * N_NODES);

    for (int i = threadIdx.x; i < N_NODES; i += blockDim.x) {
        parent[i] = i;
        posL[i]   = pos_g[f * N_NODES + i];
        ordL[i]   = order_g[f * N_NODES + i];
        deaths[i] = 0.0f;
    }
    __syncthreads();

    if (threadIdx.x == 0) {
        for (int s = 0; s < N_NODES; ++s) {
            const int v = ordL[s];
            const int ebase = v * MAX_DEG;
#pragma unroll 1
            for (int k = 0; k < MAX_DEG; ++k) {
                const int u = dst_edges[ebase + k];
                const bool valid = (u >= 0) && (posL[u] <= s);
                const int ue = valid ? u : v;
                int r1 = v;  while (parent[r1] != r1) r1 = parent[r1];
                int r2 = ue; while (parent[r2] != r2) r2 = parent[r2];
                parent[v]  = r1;                  // path compression
                parent[ue] = r2;
                if (r1 != r2) {
                    int absorbed, surv;
                    if (posL[r1] < posL[r2]) { absorbed = r2; surv = r1; }
                    else                     { absorbed = r1; surv = r2; }
                    parent[absorbed] = surv;      // younger root dies
                    deaths[posL[absorbed]] = (float)s;
                }
            }
        }
    }
    __syncthreads();

    for (int i = threadIdx.x; i < N_NODES; i += blockDim.x) {
        const int p = posL[i];
        float d = deaths[p];
        if (d == 0.0f) d = (float)N_NODES;
        pd[(size_t)i * (2 * N_FILT) + 2 * f + 0] = (float)p;
        pd[(size_t)i * (2 * N_FILT) + 2 * f + 1] = d;
    }
}

// ---------------------------------------------------------------------------
// Kernel 4: pooled[v] = pd[v] . colsum(Wt) + sum(bt)
// ---------------------------------------------------------------------------
__global__ __launch_bounds__(256) void togl_pool_kernel(
    const float* __restrict__ pd, const float* __restrict__ Wt,
    const float* __restrict__ bt, float* __restrict__ pooled)
{
    __shared__ float s[2 * N_FILT];
    __shared__ float csum;
    if (threadIdx.x < 2 * N_FILT) {
        float t = 0.0f;
        for (int h = 0; h < 64; ++h) t += Wt[h * (2 * N_FILT) + threadIdx.x];
        s[threadIdx.x] = t;
    }
    if (threadIdx.x == 2 * N_FILT) {
        float t = 0.0f;
        for (int h = 0; h < 64; ++h) t += bt[h];
        csum = t;
    }
    __syncthreads();
    const int v = blockIdx.x * 256 + threadIdx.x;
    float p = csum;
#pragma unroll
    for (int j = 0; j < 2 * N_FILT; ++j)
        p += pd[(size_t)v * (2 * N_FILT) + j] * s[j];
    pooled[v] = p;
}

// ---------------------------------------------------------------------------
// Kernel 5: out = X + pooled * Wr + br   (streaming, float4)
// ---------------------------------------------------------------------------
__global__ __launch_bounds__(256) void togl_out_kernel(
    const float* __restrict__ X, const float* __restrict__ pooled,
    const float* __restrict__ Wr, const float* __restrict__ br,
    float* __restrict__ out)
{
    const int idx = blockIdx.x * 256 + threadIdx.x;   // over N_NODES * 256
    const int v   = idx >> 8;                          // 256 float4 per row
    const int f4  = (idx & 255) << 2;
    const float p = pooled[v];
    const v4f x = *(const v4f*)(X  + (size_t)v * N_FEAT + f4);
    const v4f w = *(const v4f*)(Wr + f4);
    const v4f b = *(const v4f*)(br + f4);
    v4f o;
#pragma unroll
    for (int q = 0; q < 4; ++q) o[q] = x[q] + p * w[q] + b[q];
    *(v4f*)(out + (size_t)v * N_FEAT + f4) = o;
}

// ---------------------------------------------------------------------------
extern "C" void kernel_launch(void* const* d_in, const int* in_sizes, int n_in,
                              void* d_out, int out_size, void* d_ws, size_t ws_size,
                              hipStream_t stream)
{
    const float* X    = (const float*)d_in[0];
    const int*   edge = (const int*)d_in[1];        // (2, 65536): dst at +65536
    const float* W1   = (const float*)d_in[2];
    const float* b1   = (const float*)d_in[3];
    const float* W2   = (const float*)d_in[4];
    const float* b2   = (const float*)d_in[5];
    const float* Wt   = (const float*)d_in[6];
    const float* bt   = (const float*)d_in[7];
    const float* Wr   = (const float*)d_in[8];
    const float* br   = (const float*)d_in[9];
    float* out = (float*)d_out;

    const int* dst_edges = edge + N_NODES * MAX_DEG;

    // workspace layout
    char* ws = (char*)d_ws;
    float* filt   = (float*)ws;                                   ws += sizeof(float) * N_NODES * N_FILT;
    int*   pos    = (int*)ws;                                     ws += sizeof(int)   * N_NODES * N_FILT;
    int*   order  = (int*)ws;                                     ws += sizeof(int)   * N_NODES * N_FILT;
    float* pd     = (float*)ws;                                   ws += sizeof(float) * N_NODES * 2 * N_FILT;
    float* pooled = (float*)ws;

    // 1) filtration values (WMMA bf16)
    togl_filt_kernel<<<N_NODES / 64, 128, 0, stream>>>(X, W1, b1, W2, b2, filt);

    // 2) stable ranks per filtration
    togl_rank_kernel<<<dim3(N_NODES / 256, N_FILT), 256, 0, stream>>>(filt, pos, order);

    // 3) sequential union-find in LDS (128 KB dynamic per block)
    togl_uf_kernel<<<N_FILT, 256, 4 * N_NODES * sizeof(int), stream>>>(dst_edges, pos, order, pd);

    // 4) collapsed DeepSets pooling
    togl_pool_kernel<<<N_NODES / 256, 256, 0, stream>>>(pd, Wt, bt, pooled);

    // 5) residual output
    togl_out_kernel<<<(N_NODES * (N_FEAT / 4)) / 256, 256, 0, stream>>>(X, pooled, Wr, br, out);
}
